// HierFDI_64458869179036
// MI455X (gfx1250) — matrix operand order repaired and checked by reading, not compiled, not run on hardware
//
#include <hip/hip_runtime.h>
#include <hip/hip_bf16.h>
#include <stdint.h>

// ---------------------------------------------------------------------------
// MI455X (gfx1250) implementation.
//  * bf16 storage, fp32 accumulation via v_wmma_f32_16x16x32_bf16
//  * LSTM recurrence: 16 persistent WGs (16 sequences each); h state in LDS,
//    c state in VGPRs, Whh streamed from L2 in pre-packed WMMA-B layout with
//    pure immediate-offset addressing; fragments double-buffered.
//  * GEMMs: 2-deep software pipeline, K loop unrolled by 2 so the ping-pong
//    buffers alternate positionally (no v_mov rotation chains).
// ---------------------------------------------------------------------------

typedef __attribute__((ext_vector_type(16))) __bf16 v16bf;
typedef __attribute__((ext_vector_type(8)))  float  v8f;

struct B32x2 { uint4 lo, hi; };

// Generic fragment load: lo 16B at p, hi 16B at p+hioff elements.
// A (row-major source): hioff=16 (dw0-3 = K k..k+7, dw4-7 = K k+16..k+23).
// B (pre-packed):       hioff=8  (32 contiguous bytes per lane).
__device__ inline v16bf ldfrag(const __bf16* p, int hioff) {
    B32x2 f;
    f.lo = *(const uint4*)(p);
    f.hi = *(const uint4*)(p + hioff);
    return __builtin_bit_cast(v16bf, f);
}

__device__ inline v8f wmma_bf16(v16bf a, v16bf b, v8f c) {
    return __builtin_amdgcn_wmma_f32_16x16x32_bf16(false, a, false, b, (short)0, c, false, false);
}

// ---------------------------------------------------------------------------
// Pack W[j][k] (fp32, row-major [Nout][K]) into WMMA B-fragment layout (bf16),
// B[k][j] = W[j][k].  lane<16: K 0..7,16..23 of the 32-K tile; lane>=16: K 8..15,24..31;
// dword v holds {B[kg][col], B[kg+1][col]}, kg = (v>=4?16:0) + 8*(lane>=16) + 2*(v&3).
__global__ void pack_b_kernel(const float* __restrict__ W, __bf16* __restrict__ dst,
                              int K, int Nout) {
    int total = (K >> 5) * (Nout >> 4) * 256;
    int tid = blockIdx.x * blockDim.x + threadIdx.x;
    if (tid >= total) return;
    int v    = tid & 7;
    int lane = (tid >> 3) & 31;
    int tile = tid >> 8;
    int NT = Nout >> 4;
    int nt = tile % NT;
    int kt = tile / NT;
    int j = nt * 16 + (lane & 15);
    int k = kt * 32 + ((v & 4) ? 16 : 0) + ((lane & 16) ? 8 : 0) + 2 * (v & 3);
    size_t doff = ((size_t)(kt * NT + nt) * 32 + lane) * 16 + 2 * v;
    dst[doff]     = (__bf16)W[(size_t)j * K + k];
    dst[doff + 1] = (__bf16)W[(size_t)j * K + k + 1];
}

// x [B,T,L,D] fp32 -> xA bf16 [(b*L+l)][t][d]
__global__ void convx_kernel(const float* __restrict__ x, __bf16* __restrict__ xA,
                             int B, int T, int L, int D) {
    size_t idx = (size_t)blockIdx.x * blockDim.x + threadIdx.x;
    size_t total = (size_t)B * T * L * D;
    if (idx >= total) return;
    int d = idx % D;
    size_t r = idx / D;
    int t = r % T;
    size_t nn = r / T;
    int l = nn % L;
    int b = nn / L;
    xA[idx] = (__bf16)x[(((size_t)b * T + t) * L + l) * D + d];
}

// ---------------------------------------------------------------------------
// Generic WMMA GEMM: out[M][N] (bf16) = act( A[M][K](bf16) * B + bias (+bias2) )
// One wave computes a 16x64 output tile (4 ntiles). Requires KT = K/32 even.
// K loop unrolled by 2: buffers A/B ping-pong positionally (no copies).
__global__ __launch_bounds__(256) void gemm_bf16_kernel(
    const __bf16* __restrict__ A, const __bf16* __restrict__ Bp,
    const float* __restrict__ bias, const float* __restrict__ bias2,
    __bf16* __restrict__ out, int Mtiles, int NG, int K, int N, int relu) {
    int wid  = threadIdx.x >> 5;
    int lane = threadIdx.x & 31;
    int wg = blockIdx.x * (blockDim.x >> 5) + wid;
    if (wg >= Mtiles * NG) return;
    int mtile = wg / NG, ng = wg % NG;
    int NT = N >> 4, KT = K >> 5;
    int col  = lane & 15;
    int mrow = (lane & 16) ? 8 : 0;
    int m    = lane & 15;
    int koff = (lane & 16) ? 8 : 0;

    v8f acc[4];
#pragma unroll
    for (int it = 0; it < 4; ++it) {
        int j = (ng * 4 + it) * 16 + col;
        float bv = bias[j] + (bias2 ? bias2[j] : 0.f);
#pragma unroll
        for (int r = 0; r < 8; ++r) acc[it][r] = bv;
    }

    // fixed lane pointers; per-kt advance only
    const __bf16* pa = A + (size_t)(mtile * 16 + m) * K + koff;
    const __bf16* pb = Bp + ((size_t)(ng * 4) * 32 + lane) * 16;
    size_t bstep = (size_t)NT * 512;  // elements per kt

    v16bf aA = ldfrag(pa, 16);
    v16bf bA[4];
#pragma unroll
    for (int it = 0; it < 4; ++it) bA[it] = ldfrag(pb + it * 512, 8);
    pa += 32;
    pb += bstep;
    v16bf aB = ldfrag(pa, 16);
    v16bf bB[4];
#pragma unroll
    for (int it = 0; it < 4; ++it) bB[it] = ldfrag(pb + it * 512, 8);

    for (int kt = 0; kt < KT - 2; kt += 2) {
#pragma unroll
        for (int it = 0; it < 4; ++it) acc[it] = wmma_bf16(aA, bA[it], acc[it]);
        pa += 32;
        pb += bstep;
        aA = ldfrag(pa, 16);
#pragma unroll
        for (int it = 0; it < 4; ++it) bA[it] = ldfrag(pb + it * 512, 8);
#pragma unroll
        for (int it = 0; it < 4; ++it) acc[it] = wmma_bf16(aB, bB[it], acc[it]);
        pa += 32;
        pb += bstep;
        aB = ldfrag(pa, 16);
#pragma unroll
        for (int it = 0; it < 4; ++it) bB[it] = ldfrag(pb + it * 512, 8);
    }
#pragma unroll
    for (int it = 0; it < 4; ++it) acc[it] = wmma_bf16(aA, bA[it], acc[it]);
#pragma unroll
    for (int it = 0; it < 4; ++it) acc[it] = wmma_bf16(aB, bB[it], acc[it]);

#pragma unroll
    for (int it = 0; it < 4; ++it) {
        int j = (ng * 4 + it) * 16 + col;
#pragma unroll
        for (int r = 0; r < 8; ++r) {
            float v = acc[it][r];
            if (relu) v = v > 0.f ? v : 0.f;
            out[(size_t)(mtile * 16 + mrow + r) * N + j] = (__bf16)v;
        }
    }
}

// ---------------------------------------------------------------------------
// LSTM recurrence. One block = 16 sequences, all T steps.
// Wave w owns hcol-tiles {w, w+8} across all four gates (ntile = g*16+hc), so
// the cell update stays in-register; h lives in LDS, c in VGPRs.
// All fragment addresses are immediates off three fixed lane pointers; the
// K loop is fully unrolled (straight-line SSA, no buffer copies).
__global__ __launch_bounds__(256) void lstm_kernel(
    const __bf16* __restrict__ xp,    // [N][T][4H] gate pre-activations
    const __bf16* __restrict__ WhhP,  // packed B, K=256, N=1024 (NT=64)
    __bf16* __restrict__ hout,        // [N][T][H]
    int T) {
    const int H = 256, FH = 1024, KT = 8, LS = 264;  // LS: +8 pad vs banks
    __shared__ __bf16 h_lds[16 * LS];
    int wid  = threadIdx.x >> 5;
    int lane = threadIdx.x & 31;
    int n0   = blockIdx.x * 16;
    int col  = lane & 15;
    int mrow = (lane & 16) ? 8 : 0;
    int m    = lane & 15;
    int koff = (lane & 16) ? 8 : 0;

    for (int i = threadIdx.x; i < 16 * LS; i += 256) h_lds[i] = (__bf16)0.f;
    __syncthreads();

    v8f cst[2];
#pragma unroll
    for (int s = 0; s < 2; ++s)
#pragma unroll
        for (int r = 0; r < 8; ++r) cst[s][r] = 0.f;

    // fixed lane pointers (element units); only pseed/pout advance per t
    const __bf16* pb0     = WhhP + (size_t)wid * 512 + (size_t)lane * 16;
    const __bf16* plds_ld = h_lds + m * LS + koff;
    __bf16*       plds_st = h_lds + mrow * LS + wid * 16 + col;
    const __bf16* pseed   = xp + (size_t)(n0 + mrow) * T * FH + wid * 16 + col;
    __bf16*       pout    = hout + (size_t)(n0 + mrow) * T * H + wid * 16 + col;

    for (int t = 0; t < T; ++t) {
        // Issue xp seed loads now; consumed only after the K loop, so they
        // overlap all 64 WMMAs of this step.
        __bf16 seed[8][8];  // [g*2+s][r]
#pragma unroll
        for (int g = 0; g < 4; ++g)
#pragma unroll
            for (int s = 0; s < 2; ++s)
#pragma unroll
                for (int r = 0; r < 8; ++r)
                    seed[g * 2 + s][r] =
                        pseed[(size_t)r * T * FH + g * 256 + s * 128];

        v8f acc[8];
#pragma unroll
        for (int q = 0; q < 8; ++q)
#pragma unroll
            for (int r = 0; r < 8; ++r) acc[q][r] = 0.f;

        // fully unrolled, 2-deep pipelined K loop, all-immediate offsets
        v16bf a_c = ldfrag(plds_ld, 16);
        v16bf b_c[8];
#pragma unroll
        for (int g = 0; g < 4; ++g)
#pragma unroll
            for (int s = 0; s < 2; ++s)
                b_c[g * 2 + s] = ldfrag(pb0 + g * 8192 + s * 4096, 8);
#pragma unroll
        for (int kt = 0; kt < KT - 1; ++kt) {
            v16bf a_n = ldfrag(plds_ld + (kt + 1) * 32, 16);
            v16bf b_n[8];
#pragma unroll
            for (int g = 0; g < 4; ++g)
#pragma unroll
                for (int s = 0; s < 2; ++s)
                    b_n[g * 2 + s] =
                        ldfrag(pb0 + (kt + 1) * 32768 + g * 8192 + s * 4096, 8);
#pragma unroll
            for (int q = 0; q < 8; ++q) acc[q] = wmma_bf16(a_c, b_c[q], acc[q]);
            a_c = a_n;
#pragma unroll
            for (int q = 0; q < 8; ++q) b_c[q] = b_n[q];
        }
#pragma unroll
        for (int q = 0; q < 8; ++q) acc[q] = wmma_bf16(a_c, b_c[q], acc[q]);

        __syncthreads();  // everyone done reading previous h
#pragma unroll
        for (int s = 0; s < 2; ++s)
#pragma unroll
            for (int r = 0; r < 8; ++r) {
                float iv = 1.f / (1.f + __expf(-(acc[0 + s][r] + (float)seed[0 + s][r])));
                float fv = 1.f / (1.f + __expf(-(acc[2 + s][r] + (float)seed[2 + s][r])));
                float gv = tanhf(acc[4 + s][r] + (float)seed[4 + s][r]);
                float ov = 1.f / (1.f + __expf(-(acc[6 + s][r] + (float)seed[6 + s][r])));
                float c = fv * cst[s][r] + iv * gv;
                cst[s][r] = c;
                float h = ov * tanhf(c);
                plds_st[r * LS + s * 128] = (__bf16)h;
                pout[(size_t)r * T * H + s * 128] = (__bf16)h;
            }
        __syncthreads();  // new h visible before next step
        pseed += FH;
        pout += H;
    }
}

// ---------------------------------------------------------------------------
// Causal cumsum-weighted attention with residual; writes [B,T,L,H] bf16.
__global__ __launch_bounds__(256) void attn_kernel(
    const __bf16* __restrict__ h1, const float* __restrict__ aw,
    const float* __restrict__ ab, __bf16* __restrict__ hattn, int T, int L) {
    const int H = 256;
    __shared__ float wsh[256];
    __shared__ float earr[512];
    __shared__ float cume[512];
    __shared__ float red[256];
    int tid = threadIdx.x;
    int n = blockIdx.x, b = n / L, l = n % L;
    const __bf16* hb = h1 + (size_t)n * T * H;
    wsh[tid] = aw[tid];
    __syncthreads();
    for (int half = 0; half < 2; ++half) {
        int t = tid + half * 256;
        float s = ab[0];
        for (int h = 0; h < H; ++h) s += (float)hb[(size_t)t * H + h] * wsh[h];
        earr[t] = s;
    }
    __syncthreads();
    red[tid] = fmaxf(earr[tid], earr[tid + 256]);
    __syncthreads();
    for (int off = 128; off > 0; off >>= 1) {
        if (tid < off) red[tid] = fmaxf(red[tid], red[tid + off]);
        __syncthreads();
    }
    float smax = red[0];
    for (int half = 0; half < 2; ++half) {
        int t = tid + half * 256;
        earr[t] = __expf(earr[t] - smax);
    }
    __syncthreads();
    if (tid == 0) {
        float run = 0.f;
        for (int t = 0; t < 512; ++t) { run += earr[t]; cume[t] = run; }
    }
    __syncthreads();
    float cw = 0.f;
    for (int t = 0; t < T; ++t) {
        float hv = (float)hb[(size_t)t * H + tid];
        cw += earr[t] * hv;
        float o = hv + cw / (cume[t] + 1e-12f);
        hattn[(((size_t)b * T + t) * L + l) * H + tid] = (__bf16)o;
    }
}

// LayerNorm over H=256, one wave per row.
__global__ __launch_bounds__(256) void ln_kernel(
    const __bf16* __restrict__ src, const float* __restrict__ g,
    const float* __restrict__ bb, __bf16* __restrict__ dst, int rows) {
    int wid = threadIdx.x >> 5, lane = threadIdx.x & 31;
    int row = blockIdx.x * 8 + wid;
    if (row >= rows) return;
    const __bf16* p = src + (size_t)row * 256;
    float x[8], sum = 0.f;
#pragma unroll
    for (int i = 0; i < 8; ++i) { x[i] = (float)p[lane + 32 * i]; sum += x[i]; }
    for (int o = 16; o > 0; o >>= 1) sum += __shfl_xor(sum, o, 32);
    float mean = sum * (1.f / 256.f);
    float vs = 0.f;
#pragma unroll
    for (int i = 0; i < 8; ++i) { float d = x[i] - mean; vs += d * d; }
    for (int o = 16; o > 0; o >>= 1) vs += __shfl_xor(vs, o, 32);
    float inv = rsqrtf(vs * (1.f / 256.f) + 1e-5f);
#pragma unroll
    for (int i = 0; i < 8; ++i) {
        int h = lane + 32 * i;
        dst[(size_t)row * 256 + h] = (__bf16)((x[i] - mean) * inv * g[h] + bb[h]);
    }
}

// rank-1 head: out[(row/Ldiv)*ostride + obase + row%Ldiv] = dot(X[row], w)+b
__global__ __launch_bounds__(256) void dot_head_kernel(
    const __bf16* __restrict__ X, int Kdim, const float* __restrict__ w,
    const float* __restrict__ bias, float* __restrict__ out,
    int Ldiv, int ostride, int obase, int rows) {
    int wid = threadIdx.x >> 5, lane = threadIdx.x & 31;
    int row = blockIdx.x * 8 + wid;
    if (row >= rows) return;
    const __bf16* p = X + (size_t)row * Kdim;
    float acc = 0.f;
    for (int k = lane; k < Kdim; k += 32) acc += (float)p[k] * w[k];
    for (int o = 16; o > 0; o >>= 1) acc += __shfl_xor(acc, o, 32);
    if (lane == 0)
        out[(size_t)(row / Ldiv) * ostride + obase + (row % Ldiv)] = acc + bias[0];
}

// tiny dense layer (Nout <= 8): thread per (row, j)
__global__ void fc_small_kernel(const __bf16* __restrict__ X, int Kdim,
                                const float* __restrict__ W, const float* __restrict__ b,
                                float* __restrict__ out, int Nout, int rows) {
    size_t idx = (size_t)blockIdx.x * blockDim.x + threadIdx.x;
    if (idx >= (size_t)rows * Nout) return;
    size_t row = idx / Nout;
    int j = idx % Nout;
    const __bf16* p = X + row * Kdim;
    float acc = b[j];
    for (int k = 0; k < Kdim; ++k) acc += (float)p[k] * W[j * Kdim + k];
    out[row * Nout + j] = acc;
}

// ---------------------------------------------------------------------------
extern "C" void kernel_launch(void* const* d_in, const int* in_sizes, int n_in,
                              void* d_out, int out_size, void* d_ws, size_t ws_size,
                              hipStream_t stream) {
    (void)in_sizes; (void)n_in; (void)out_size; (void)ws_size;
    const int B = 16, T = 512, L = 16, D = 64, H = 256;
    const int N = B * L;         // 256 sequences
    const int ROWS = N * T;      // 131072
    const int BT = B * T;        // 8192
    const int FH = 4 * H;        // 1024

    const float* x       = (const float*)d_in[0];
    const float* Wih0    = (const float*)d_in[1];
    const float* Whh0    = (const float*)d_in[2];
    const float* bih0    = (const float*)d_in[3];
    const float* bhh0    = (const float*)d_in[4];
    const float* Wih1    = (const float*)d_in[5];
    const float* Whh1    = (const float*)d_in[6];
    const float* bih1    = (const float*)d_in[7];
    const float* bhh1    = (const float*)d_in[8];
    const float* attn_w  = (const float*)d_in[9];
    const float* attn_b  = (const float*)d_in[10];
    const float* link_w  = (const float*)d_in[11];
    const float* link_b  = (const float*)d_in[12];
    const float* nof_w1  = (const float*)d_in[13];
    const float* nof_b1  = (const float*)d_in[14];
    const float* nof_w2  = (const float*)d_in[15];
    const float* nof_b2  = (const float*)d_in[16];
    const float* ln_g    = (const float*)d_in[17];
    const float* ln_b    = (const float*)d_in[18];
    const float* reg_w1  = (const float*)d_in[19];
    const float* reg_b1  = (const float*)d_in[20];
    const float* reg_w2  = (const float*)d_in[21];
    const float* reg_b2  = (const float*)d_in[22];
    const float* mot_w1  = (const float*)d_in[23];
    const float* mot_b1  = (const float*)d_in[24];
    const float* mot_w2  = (const float*)d_in[25];
    const float* mot_b2  = (const float*)d_in[26];

    uint8_t* ws = (uint8_t*)d_ws;
    size_t off = 0;
    auto alloc = [&](size_t bytes) { void* p = ws + off; off += bytes; return p; };

    __bf16* xA     = (__bf16*)alloc((size_t)N * T * D * 2);        // 16.8 MB
    __bf16* wih0p  = (__bf16*)alloc((size_t)D * FH * 2);
    __bf16* whh0p  = (__bf16*)alloc((size_t)H * FH * 2);
    __bf16* wih1p  = (__bf16*)alloc((size_t)H * FH * 2);
    __bf16* whh1p  = (__bf16*)alloc((size_t)H * FH * 2);
    __bf16* nofw1p = (__bf16*)alloc((size_t)L * H * 256 * 2);      // K=4096,N=256
    __bf16* regw1p = (__bf16*)alloc((size_t)H * H * 2);
    __bf16* motw1p = (__bf16*)alloc((size_t)H * 128 * 2);
    __bf16* h0     = (__bf16*)alloc((size_t)N * T * H * 2);        // 67 MB
    __bf16* h1     = (__bf16*)alloc((size_t)N * T * H * 2);        // 67 MB
    __bf16* xp     = (__bf16*)alloc((size_t)N * T * FH * 2);       // 268 MB
    // After LSTM layer 1, xp is dead -> alias head buffers into it.
    __bf16* hattn = xp;                                            // [B,T,L,H] 67 MB
    __bf16* hln   = hattn + (size_t)ROWS * H;                      // 67 MB
    __bf16* regh  = hln + (size_t)ROWS * H;                        // 67 MB
    __bf16* moth  = regh + (size_t)ROWS * H;                       // 33.5 MB
    __bf16* nofh  = moth + (size_t)ROWS * 128;                     // 4.2 MB

    float* out = (float*)d_out;
    float* out_l1  = out;                         // [B,T,17]
    float* out_reg = out + (size_t)BT * 17;       // [B,T,L,4]
    float* out_mot = out_reg + (size_t)ROWS * 4;  // [B,T,L,8]

    // 1. x -> bf16, reorder to [(b,l)][t][d]
    {
        size_t total = (size_t)N * T * D;
        convx_kernel<<<(int)((total + 255) / 256), 256, 0, stream>>>(x, xA, B, T, L, D);
    }
    // 2. pack all WMMA B operands
    auto pack = [&](const float* W, __bf16* dst, int K, int Nout) {
        int total = (K >> 5) * (Nout >> 4) * 256;
        pack_b_kernel<<<(total + 255) / 256, 256, 0, stream>>>(W, dst, K, Nout);
    };
    pack(Wih0, wih0p, D, FH);
    pack(Whh0, whh0p, H, FH);
    pack(Wih1, wih1p, H, FH);
    pack(Whh1, whh1p, H, FH);
    pack(nof_w1, nofw1p, L * H, 256);
    pack(reg_w1, regw1p, H, H);
    pack(mot_w1, motw1p, H, 128);

    // 3. layer-0 input projection  xp = xA * Wih0^T + bih0 + bhh0
    gemm_bf16_kernel<<<(ROWS / 16) * (FH / 64) / 8, 256, 0, stream>>>(
        xA, wih0p, bih0, bhh0, xp, ROWS / 16, FH / 64, D, FH, 0);
    // 4. layer-0 recurrence
    lstm_kernel<<<N / 16, 256, 0, stream>>>(xp, whh0p, h0, T);
    // 5. layer-1 input projection  xp = h0 * Wih1^T + bih1 + bhh1
    gemm_bf16_kernel<<<(ROWS / 16) * (FH / 64) / 8, 256, 0, stream>>>(
        h0, wih1p, bih1, bhh1, xp, ROWS / 16, FH / 64, H, FH, 0);
    // 6. layer-1 recurrence
    lstm_kernel<<<N / 16, 256, 0, stream>>>(xp, whh1p, h1, T);
    // 7. attention (+residual), transpose to [B,T,L,H]
    attn_kernel<<<N, 256, 0, stream>>>(h1, attn_w, attn_b, hattn, T, L);
    // 8. LayerNorm for region head
    ln_kernel<<<ROWS / 8, 256, 0, stream>>>(hattn, ln_g, ln_b, hln, ROWS);
    // 9. head hidden layers (WMMA + relu)
    gemm_bf16_kernel<<<(BT / 16) * (256 / 64) / 8, 256, 0, stream>>>(
        hattn, nofw1p, nof_b1, nullptr, nofh, BT / 16, 256 / 64, L * H, 256, 1);
    gemm_bf16_kernel<<<(ROWS / 16) * (256 / 64) / 8, 256, 0, stream>>>(
        hln, regw1p, reg_b1, nullptr, regh, ROWS / 16, 256 / 64, H, 256, 1);
    gemm_bf16_kernel<<<(ROWS / 16) * (128 / 64) / 8, 256, 0, stream>>>(
        hattn, motw1p, mot_b1, nullptr, moth, ROWS / 16, 128 / 64, H, 128, 1);
    // 10. output layers
    dot_head_kernel<<<BT / 8, 256, 0, stream>>>(nofh, 256, nof_w2, nof_b2,
                                                out_l1, 1, 17, 0, BT);
    dot_head_kernel<<<ROWS / 8, 256, 0, stream>>>(hattn, 256, link_w, link_b,
                                                  out_l1, L, 17, 1, ROWS);
    fc_small_kernel<<<(ROWS * 4 + 255) / 256, 256, 0, stream>>>(
        regh, 256, reg_w2, reg_b2, out_reg, 4, ROWS);
    fc_small_kernel<<<(ROWS * 8 + 255) / 256, 256, 0, stream>>>(
        moth, 128, mot_w2, mot_b2, out_mot, 8, ROWS);
}
